// MamboPoolingWithClusterAttention_81320910782701
// MI455X (gfx1250) — compile-verified, matrix operand order checked by reading, not run on hardware
//
#include <hip/hip_runtime.h>
#include <hip/hip_bf16.h>

// Problem constants (from reference)
#define NN   4096
#define BB   8
#define DD   256
#define HH   4
#define NPGC 512
#define EE   65536
#define KTOP 103            // ceil(512*0.2)
#define CC   (BB*KTOP)      // 824
#define EF   (EE+NN)        // edges + self loops
#define NEGF (-1e30f)

typedef __attribute__((ext_vector_type(16))) __bf16 v16bf;
typedef __attribute__((ext_vector_type(8)))  float  v8f;

union Frag { v16bf v; uint4 u[2]; };

// ---------- helpers ----------
__device__ __forceinline__ unsigned short f2bf_rne(float f) {
    unsigned u = __float_as_uint(f);
    unsigned lsb = (u >> 16) & 1u;
    u += 0x7fffu + lsb;
    return (unsigned short)(u >> 16);
}
// monotone float<->uint key for atomic float max
__device__ __forceinline__ unsigned fkey(float f) {
    unsigned u = __float_as_uint(f);
    return (u & 0x80000000u) ? ~u : (u | 0x80000000u);
}
__device__ __forceinline__ float kinv(unsigned k) {
    return (k & 0x80000000u) ? __uint_as_float(k & 0x7fffffffu)
                             : __uint_as_float(~k);
}

// ---------- fills / conversions ----------
__global__ void k_fill_f32(float* p, float v, int n) {
    int i = blockIdx.x * blockDim.x + threadIdx.x; if (i < n) p[i] = v;
}
__global__ void k_fill_u32(unsigned* p, unsigned v, int n) {
    int i = blockIdx.x * blockDim.x + threadIdx.x; if (i < n) p[i] = v;
}
__global__ void k_fill_i32(int* p, int v, int n) {
    int i = blockIdx.x * blockDim.x + threadIdx.x; if (i < n) p[i] = v;
}
__global__ void k_f2bf(const float* __restrict__ s, unsigned short* __restrict__ d, int n) {
    int i = blockIdx.x * blockDim.x + threadIdx.x; if (i < n) d[i] = f2bf_rne(s[i]);
}

// ---------- gf = segment_mean(x, batch, B); batch is contiguous blocks of 512 ----------
__global__ void k_gf(const float* __restrict__ x, float* __restrict__ gf) {
    int i = blockIdx.x * blockDim.x + threadIdx.x;
    if (i < BB * DD) {
        int b = i / DD, d = i % DD;
        const float* p = x + (size_t)b * NPGC * DD + d;
        float s = 0.f;
        for (int t = 0; t < NPGC; ++t) s += p[(size_t)t * DD];
        gf[i] = s * (1.0f / NPGC);
    }
}

// ---------- bf16 WMMA GEMM: C[M,Nn] = A[M,K] @ W[Nn,K]^T (+bias[Nn]) ----------
// one wave -> 16x64 output strip (4 WMMA tiles sharing the A fragment)
__global__ void k_gemm_bf16(const unsigned short* __restrict__ A,
                            const unsigned short* __restrict__ W,
                            const float* __restrict__ bias,
                            float* __restrict__ Cm,
                            int M, int Nn, int Kk) {
    int wid    = (blockIdx.x * blockDim.x + threadIdx.x) >> 5;
    int tilesN = Nn >> 6;
    int nTiles = (M >> 4) * tilesN;
    if (wid >= nTiles) return;                  // uniform per wave: EXEC stays all-ones
    int tm   = (wid / tilesN) << 4;
    int tn   = (wid % tilesN) << 6;
    int lane = threadIdx.x & 31;
    int row  = lane & 15;                        // M (or N) index within tile
    int kb   = (lane >> 4) << 3;                 // K half select per ISA 16-bit layout

    const unsigned short* arow = A + (size_t)(tm + row) * Kk + kb;
    const unsigned short* w0   = W + (size_t)(tn + row) * Kk + kb;
    const unsigned short* w1   = w0 + (size_t)16 * Kk;
    const unsigned short* w2   = w0 + (size_t)32 * Kk;
    const unsigned short* w3   = w0 + (size_t)48 * Kk;

    v8f acc0 = {}; v8f acc1 = {}; v8f acc2 = {}; v8f acc3 = {};
    for (int k0 = 0; k0 < Kk; k0 += 32) {
        Frag fa, f0, f1, f2, f3;
        fa.u[0] = *(const uint4*)(arow + k0); fa.u[1] = *(const uint4*)(arow + k0 + 16);
        f0.u[0] = *(const uint4*)(w0 + k0);   f0.u[1] = *(const uint4*)(w0 + k0 + 16);
        f1.u[0] = *(const uint4*)(w1 + k0);   f1.u[1] = *(const uint4*)(w1 + k0 + 16);
        f2.u[0] = *(const uint4*)(w2 + k0);   f2.u[1] = *(const uint4*)(w2 + k0 + 16);
        f3.u[0] = *(const uint4*)(w3 + k0);   f3.u[1] = *(const uint4*)(w3 + k0 + 16);
        acc0 = __builtin_amdgcn_wmma_f32_16x16x32_bf16(false, fa.v, false, f0.v, (short)0, acc0, false, false);
        acc1 = __builtin_amdgcn_wmma_f32_16x16x32_bf16(false, fa.v, false, f1.v, (short)0, acc1, false, false);
        acc2 = __builtin_amdgcn_wmma_f32_16x16x32_bf16(false, fa.v, false, f2.v, (short)0, acc2, false, false);
        acc3 = __builtin_amdgcn_wmma_f32_16x16x32_bf16(false, fa.v, false, f3.v, (short)0, acc3, false, false);
    }
    int rbase = tm + ((lane >> 4) << 3);         // C/D layout: lanes16-31 hold M+8
    int cb    = tn + row;
    float b0 = bias ? bias[cb]      : 0.f;
    float b1 = bias ? bias[cb + 16] : 0.f;
    float b2 = bias ? bias[cb + 32] : 0.f;
    float b3 = bias ? bias[cb + 48] : 0.f;
#pragma unroll
    for (int r = 0; r < 8; ++r) {
        float* crow = Cm + (size_t)(rbase + r) * Nn;
        crow[cb]      = acc0[r] + b0;
        crow[cb + 16] = acc1[r] + b1;
        crow[cb + 32] = acc2[r] + b2;
        crow[cb + 48] = acc3[r] + b3;
    }
}

// ---------- kb, vb: tiny 8-row projections of gf (k and v have only B distinct rows) ----------
__global__ void k_kv(const float* __restrict__ gf, const float* __restrict__ w,
                     const float* __restrict__ bias, float* __restrict__ kb, float* __restrict__ vb) {
    int i = blockIdx.x * blockDim.x + threadIdx.x;
    if (i < BB * 2 * DD) {
        int b = i / (2 * DD); int j = i % (2 * DD);
        int sel = j / DD;     int d = j % DD;
        int rowi = DD + sel * DD + d;                 // Wk rows [D,2D), Wv rows [2D,3D)
        const float* wr = w + (size_t)rowi * DD;
        const float* g  = gf + (size_t)b * DD;
        float s = bias[rowi];
        for (int k = 0; k < DD; ++k) s += g[k] * wr[k];
        (sel ? vb : kb)[b * DD + d] = s;
    }
}

// ---------- attention collapsed to 8 distinct keys per node; writes o in bf16 for out_proj GEMM ----------
__global__ void k_attn(const float* __restrict__ q, const float* __restrict__ kb,
                       const float* __restrict__ vb, unsigned short* __restrict__ obf) {
    int n = blockIdx.x; int t = threadIdx.x; int h = t >> 6;
    __shared__ float s_s[HH * BB];
    __shared__ float s_p[HH * BB];
    if (t < HH * BB) s_s[t] = 0.f;
    __syncthreads();
    float qv = q[(size_t)n * DD + t];
    for (int b = 0; b < BB; ++b) atomicAdd(&s_s[h * BB + b], qv * kb[b * DD + t]);
    __syncthreads();
    if (t < HH) {
        float vv[BB]; float m = NEGF;
        for (int b = 0; b < BB; ++b) { vv[b] = s_s[t * BB + b] * 0.125f; m = fmaxf(m, vv[b]); }
        float sum = 0.f;
        for (int b = 0; b < BB; ++b) { vv[b] = __expf(vv[b] - m); sum += vv[b]; }
        float inv = 1.f / sum;
        for (int b = 0; b < BB; ++b) s_p[t * BB + b] = vv[b] * inv;
    }
    __syncthreads();
    float ov = 0.f;
    for (int b = 0; b < BB; ++b) ov += s_p[h * BB + b] * vb[b * DD + t];
    obf[(size_t)n * DD + t] = f2bf_rne(ov);
}

// ---------- score = cluster_attn row sums ----------
__global__ void k_score(const float* __restrict__ cattn, float* __restrict__ score) {
    int i = blockIdx.x * blockDim.x + threadIdx.x;
    if (i < NN) {
        const float* p = cattn + (size_t)i * DD;
        float s = 0.f;
        for (int d = 0; d < DD; ++d) s += p[d];
        score[i] = s;
    }
}

// ---------- per-batch top-K (iterative LDS argmax, tie -> lowest index like jax top_k) ----------
__global__ void k_topk(const float* __restrict__ score, int* __restrict__ centers, int* __restrict__ cid) {
    int b = blockIdx.x; int t = threadIdx.x;
    __shared__ float sv[NPGC];
    __shared__ float rv[NPGC];
    __shared__ int   ri[NPGC];
    sv[t] = score[b * NPGC + t];
    __syncthreads();
    for (int it = 0; it < KTOP; ++it) {
        rv[t] = sv[t]; ri[t] = t; __syncthreads();
        for (int s = NPGC / 2; s > 0; s >>= 1) {
            if (t < s) {
                float v2 = rv[t + s]; int i2 = ri[t + s];
                if (v2 > rv[t] || (v2 == rv[t] && i2 < ri[t])) { rv[t] = v2; ri[t] = i2; }
            }
            __syncthreads();
        }
        if (t == 0) {
            int g = ri[0];
            centers[b * KTOP + it] = b * NPGC + g;
            cid[b * NPGC + g] = b * KTOP + it;
            sv[g] = NEGF;
        }
        __syncthreads();
    }
}

// ---------- GAT attention scalars a_s, a_d ----------
__global__ void k_att_scores(const float* __restrict__ xW, const float* __restrict__ atts,
                             const float* __restrict__ attd, float* __restrict__ a_s, float* __restrict__ a_d) {
    int i = blockIdx.x * blockDim.x + threadIdx.x;
    if (i < NN * HH) {
        int n = i >> 2, h = i & 3;
        const float* xp = xW + (size_t)n * (HH * DD) + h * DD;
        const float* as = atts + h * DD;
        const float* ad = attd + h * DD;
        float s1 = 0.f, s2 = 0.f;
        for (int d = 0; d < DD; ++d) { float v = xp[d]; s1 += v * as[d]; s2 += v * ad[d]; }
        a_s[i] = s1; a_d[i] = s2;
    }
}

// ---------- GAT edge softmax over dst (3-phase with atomic key-max) ----------
__global__ void k_alpha1(const int* __restrict__ src, const int* __restrict__ dst,
                         const float* __restrict__ a_s, const float* __restrict__ a_d,
                         float* __restrict__ alpha, unsigned* __restrict__ amaxk) {
    int j = blockIdx.x * blockDim.x + threadIdx.x;
    if (j < EF) {
        int s  = (j < EE) ? src[j] : (j - EE);
        int dd = (j < EE) ? dst[j] : (j - EE);
        for (int h = 0; h < HH; ++h) {
            float v = a_s[s * HH + h] + a_d[dd * HH + h];
            v = (v > 0.f) ? v : 0.2f * v;                      // leaky_relu(0.2)
            alpha[(size_t)j * HH + h] = v;
            atomicMax(&amaxk[dd * HH + h], fkey(v));
        }
    }
}
__global__ void k_alpha2(const int* __restrict__ src, const int* __restrict__ dst,
                         float* __restrict__ alpha, const unsigned* __restrict__ amaxk,
                         float* __restrict__ asum) {
    int j = blockIdx.x * blockDim.x + threadIdx.x;
    if (j < EF) {
        int dd = (j < EE) ? dst[j] : (j - EE);
        for (int h = 0; h < HH; ++h) {
            float ex = __expf(alpha[(size_t)j * HH + h] - kinv(amaxk[dd * HH + h]));
            alpha[(size_t)j * HH + h] = ex;
            atomicAdd(&asum[dd * HH + h], ex);
        }
    }
}
__global__ void k_alpha3(const int* __restrict__ src, const int* __restrict__ dst,
                         float* __restrict__ alpha, const float* __restrict__ asum,
                         const float* __restrict__ wred, const float* __restrict__ bred,
                         float* __restrict__ e, unsigned* __restrict__ emaxk) {
    int j = blockIdx.x * blockDim.x + threadIdx.x;
    if (j < EF) {
        int dd = (j < EE) ? dst[j] : (j - EE);
        float af[HH];
        for (int h = 0; h < HH; ++h) {
            float a = alpha[(size_t)j * HH + h] / asum[dd * HH + h];
            alpha[(size_t)j * HH + h] = a;
            af[h] = a;
        }
        if (j < EE) {                                         // e = e_full[:E]
            float s = bred[0];
            for (int h = 0; h < HH; ++h) s += af[h] * wred[h];
            e[j] = s;
            atomicMax(&emaxk[dst[j]], fkey(s));
        }
    }
}
__global__ void k_edge_exp(const int* __restrict__ dst, float* __restrict__ e,
                           const unsigned* __restrict__ emaxk, float* __restrict__ esum) {
    int j = blockIdx.x * blockDim.x + threadIdx.x;
    if (j < EE) {
        float ex = __expf(e[j] - kinv(emaxk[dst[j]]));
        e[j] = ex;
        atomicAdd(&esum[dst[j]], ex);
    }
}
__global__ void k_edge_norm(const int* __restrict__ dst, const float* __restrict__ e,
                            const float* __restrict__ esum, float* __restrict__ escore) {
    int j = blockIdx.x * blockDim.x + threadIdx.x;
    if (j < EE) escore[j] = e[j] / esum[dst[j]];
}

// ---------- GAT aggregation fused with mean over heads ----------
__global__ void k_gat_scatter(const int* __restrict__ src, const int* __restrict__ dst,
                              const float* __restrict__ alpha, const float* __restrict__ xW,
                              float* __restrict__ xatt) {
    int j = blockIdx.x; int d = threadIdx.x;
    int s  = (j < EE) ? src[j] : (j - EE);
    int dd = (j < EE) ? dst[j] : (j - EE);
    const float* xp = xW + (size_t)s * (HH * DD) + d;
    float a0 = alpha[(size_t)j * HH + 0];
    float a1 = alpha[(size_t)j * HH + 1];
    float a2 = alpha[(size_t)j * HH + 2];
    float a3 = alpha[(size_t)j * HH + 3];
    float tmp = a0 * xp[0] + a1 * xp[DD] + a2 * xp[2 * DD] + a3 * xp[3 * DD];
    atomicAdd(&xatt[(size_t)dd * DD + d], tmp * 0.25f);
}

// ---------- x2 = x + x_attned + mean_h(gat_bias) ----------
__global__ void k_x2(const float* __restrict__ x, const float* __restrict__ xatt,
                     const float* __restrict__ gb, float* __restrict__ x2) {
    int i = blockIdx.x * blockDim.x + threadIdx.x;
    if (i < NN * DD) {
        int d = i & (DD - 1);
        float bm = 0.25f * (gb[d] + gb[DD + d] + gb[2 * DD + d] + gb[3 * DD + d]);
        x2[i] = x[i] + xatt[i] + bm;
    }
}

// ---------- cluster assignment via best-scoring center neighbor ----------
__global__ void k_best1(const int* __restrict__ src, const int* __restrict__ dst,
                        const float* __restrict__ escore, const int* __restrict__ cid,
                        unsigned* __restrict__ bestk) {
    int j = blockIdx.x * blockDim.x + threadIdx.x;
    if (j < 2 * EE) {
        int e0 = (j < EE) ? j : (j - EE);
        int u  = (j < EE) ? src[e0] : dst[e0];
        int nb = (j < EE) ? dst[e0] : src[e0];
        if (cid[nb] >= 0) atomicMax(&bestk[u], fkey(escore[e0]));
    }
}
__global__ void k_best2(const int* __restrict__ src, const int* __restrict__ dst,
                        const float* __restrict__ escore, const int* __restrict__ cid,
                        const unsigned* __restrict__ bestk, int* __restrict__ nbc) {
    int j = blockIdx.x * blockDim.x + threadIdx.x;
    if (j < 2 * EE) {
        int e0 = (j < EE) ? j : (j - EE);
        int u  = (j < EE) ? src[e0] : dst[e0];
        int nb = (j < EE) ? dst[e0] : src[e0];
        int c = cid[nb];
        if (c >= 0 && fkey(escore[e0]) >= bestk[u]) atomicMax(&nbc[u], c);
    }
}
__global__ void k_assign(const int* __restrict__ cid, const int* __restrict__ nbc,
                         const int* __restrict__ batch, int* __restrict__ assign) {
    int n = blockIdx.x * blockDim.x + threadIdx.x;
    if (n < NN) {
        int c = cid[n];
        assign[n] = (c >= 0) ? c : ((nbc[n] >= 0) ? nbc[n] : batch[n] * KTOP);
    }
}

// ---------- pooled mean + outputs ----------
__global__ void k_pool(const int* __restrict__ assign, const float* __restrict__ x2,
                       float* __restrict__ psum, float* __restrict__ pcnt) {
    int n = blockIdx.x; int d = threadIdx.x;
    int a = assign[n];
    atomicAdd(&psum[(size_t)a * DD + d], x2[(size_t)n * DD + d]);
    if (d == 0) atomicAdd(&pcnt[a], 1.0f);
}
__global__ void k_poolout(const float* __restrict__ psum, const float* __restrict__ pcnt,
                          float* __restrict__ out) {
    int i = blockIdx.x * blockDim.x + threadIdx.x;
    if (i < CC * DD) out[i] = psum[i] / fmaxf(pcnt[i / DD], 1.0f);
}
__global__ void k_edgeout(const int* __restrict__ src, const int* __restrict__ dst,
                          const int* __restrict__ assign, float* __restrict__ out) {
    int j = blockIdx.x * blockDim.x + threadIdx.x;
    if (j < 2 * EE) {
        int v = (j < EE) ? assign[src[j]] : assign[dst[j - EE]];
        out[(size_t)CC * DD + j] = (float)v;
    }
}
__global__ void k_batchout(float* __restrict__ out) {
    int c = blockIdx.x * blockDim.x + threadIdx.x;
    if (c < CC) out[(size_t)CC * DD + 2 * (size_t)EE + c] = (float)(c / KTOP);
}

// ====================== host ======================
extern "C" void kernel_launch(void* const* d_in, const int* in_sizes, int n_in,
                              void* d_out, int out_size, void* d_ws, size_t ws_size,
                              hipStream_t stream) {
    (void)in_sizes; (void)n_in; (void)out_size; (void)ws_size;
    const float* x     = (const float*)d_in[0];
    const int*   ei    = (const int*)d_in[1];
    const int*   src   = ei;
    const int*   dst   = ei + EE;
    const int*   batch = (const int*)d_in[2];
    const float* ipw   = (const float*)d_in[3];
    const float* ipb   = (const float*)d_in[4];
    const float* opw   = (const float*)d_in[5];
    const float* opb   = (const float*)d_in[6];
    const float* wgat  = (const float*)d_in[7];
    const float* atts  = (const float*)d_in[8];
    const float* attd  = (const float*)d_in[9];
    const float* gbias = (const float*)d_in[10];
    const float* wred  = (const float*)d_in[11];
    const float* bred  = (const float*)d_in[12];
    float* out = (float*)d_out;

    // bump allocator over d_ws
    char* wsp = (char*)d_ws; size_t off = 0;
    auto alloc = [&](size_t bytes) -> void* {
        void* p = wsp + off;
        off = (off + bytes + 255) & ~(size_t)255;
        return p;
    };
    unsigned short* xbf   = (unsigned short*)alloc((size_t)NN * DD * 2);
    unsigned short* wqbf  = (unsigned short*)alloc((size_t)DD * DD * 2);
    unsigned short* wobf  = (unsigned short*)alloc((size_t)DD * DD * 2);
    unsigned short* wgbf  = (unsigned short*)alloc((size_t)HH * DD * DD * 2);
    unsigned short* obf   = (unsigned short*)alloc((size_t)NN * DD * 2);
    float* gf    = (float*)alloc((size_t)BB * DD * 4);
    float* q     = (float*)alloc((size_t)NN * DD * 4);
    float* kb    = (float*)alloc((size_t)BB * DD * 4);
    float* vb    = (float*)alloc((size_t)BB * DD * 4);
    float* cattn = (float*)alloc((size_t)NN * DD * 4);
    float* xW    = (float*)alloc((size_t)NN * HH * DD * 4);
    float* score = (float*)alloc((size_t)NN * 4);
    int*   centers = (int*)alloc((size_t)CC * 4);
    int*   cid     = (int*)alloc((size_t)NN * 4);
    float* a_s   = (float*)alloc((size_t)NN * HH * 4);
    float* a_d   = (float*)alloc((size_t)NN * HH * 4);
    float* alpha = (float*)alloc((size_t)EF * HH * 4);
    unsigned* amaxk = (unsigned*)alloc((size_t)NN * HH * 4);
    float* asum  = (float*)alloc((size_t)NN * HH * 4);
    float* e     = (float*)alloc((size_t)EE * 4);
    unsigned* emaxk = (unsigned*)alloc((size_t)NN * 4);
    float* esum  = (float*)alloc((size_t)NN * 4);
    float* escore = (float*)alloc((size_t)EE * 4);
    float* xatt  = (float*)alloc((size_t)NN * DD * 4);
    float* x2    = (float*)alloc((size_t)NN * DD * 4);
    unsigned* bestk = (unsigned*)alloc((size_t)NN * 4);
    int*   nbc   = (int*)alloc((size_t)NN * 4);
    int*   assign = (int*)alloc((size_t)NN * 4);
    float* psum  = (float*)alloc((size_t)CC * DD * 4);
    float* pcnt  = (float*)alloc((size_t)CC * 4);

    const int T = 256;
    auto nb = [](int n, int t) { return (n + t - 1) / t; };
    auto gemm = [&](const unsigned short* A, const unsigned short* Wm, const float* bias,
                    float* Cm, int M, int Nn, int Kk) {
        int tiles = (M / 16) * (Nn / 64);
        k_gemm_bf16<<<nb(tiles, 8), 256, 0, stream>>>(A, Wm, bias, Cm, M, Nn, Kk);
    };

    // conversions + global features
    k_f2bf<<<nb(NN * DD, T), T, 0, stream>>>(x, xbf, NN * DD);
    k_f2bf<<<nb(DD * DD, T), T, 0, stream>>>(ipw, wqbf, DD * DD);             // Wq = in_proj rows [0,D)
    k_f2bf<<<nb(DD * DD, T), T, 0, stream>>>(opw, wobf, DD * DD);
    k_f2bf<<<nb(HH * DD * DD, T), T, 0, stream>>>(wgat, wgbf, HH * DD * DD);
    k_gf<<<nb(BB * DD, T), T, 0, stream>>>(x, gf);

    // q projection (WMMA), tiny k/v projections (8 distinct rows)
    gemm(xbf, wqbf, ipb, q, NN, DD, DD);
    k_kv<<<nb(BB * 2 * DD, T), T, 0, stream>>>(gf, ipw, ipb, kb, vb);

    // collapsed attention + out projection (WMMA)
    k_attn<<<NN, 256, 0, stream>>>(q, kb, vb, obf);
    gemm(obf, wobf, opb, cattn, NN, DD, DD);

    // GAT projection (WMMA)
    gemm(xbf, wgbf, nullptr, xW, NN, HH * DD, DD);

    // top-k centers
    k_score<<<nb(NN, T), T, 0, stream>>>(cattn, score);
    k_fill_i32<<<nb(NN, T), T, 0, stream>>>(cid, -1, NN);
    k_topk<<<BB, NPGC, 0, stream>>>(score, centers, cid);

    // GAT attention
    k_att_scores<<<nb(NN * HH, T), T, 0, stream>>>(xW, atts, attd, a_s, a_d);
    k_fill_u32<<<nb(NN * HH, T), T, 0, stream>>>(amaxk, 0u, NN * HH);
    k_fill_f32<<<nb(NN * HH, T), T, 0, stream>>>(asum, 0.f, NN * HH);
    k_alpha1<<<nb(EF, T), T, 0, stream>>>(src, dst, a_s, a_d, alpha, amaxk);
    k_alpha2<<<nb(EF, T), T, 0, stream>>>(src, dst, alpha, amaxk, asum);
    k_fill_u32<<<nb(NN, T), T, 0, stream>>>(emaxk, 0u, NN);
    k_fill_f32<<<nb(NN, T), T, 0, stream>>>(esum, 0.f, NN);
    k_alpha3<<<nb(EF, T), T, 0, stream>>>(src, dst, alpha, asum, wred, bred, e, emaxk);
    k_edge_exp<<<nb(EE, T), T, 0, stream>>>(dst, e, emaxk, esum);
    k_edge_norm<<<nb(EE, T), T, 0, stream>>>(dst, e, esum, escore);

    // GAT aggregation + residual
    k_fill_f32<<<nb(NN * DD, T), T, 0, stream>>>(xatt, 0.f, NN * DD);
    k_gat_scatter<<<EF, DD, 0, stream>>>(src, dst, alpha, xW, xatt);
    k_x2<<<nb(NN * DD, T), T, 0, stream>>>(x, xatt, gbias, x2);

    // cluster assignment
    k_fill_u32<<<nb(NN, T), T, 0, stream>>>(bestk, 0u, NN);
    k_fill_i32<<<nb(NN, T), T, 0, stream>>>(nbc, -1, NN);
    k_best1<<<nb(2 * EE, T), T, 0, stream>>>(src, dst, escore, cid, bestk);
    k_best2<<<nb(2 * EE, T), T, 0, stream>>>(src, dst, escore, cid, bestk, nbc);
    k_assign<<<nb(NN, T), T, 0, stream>>>(cid, nbc, batch, assign);

    // pooled outputs
    k_fill_f32<<<nb(CC * DD, T), T, 0, stream>>>(psum, 0.f, CC * DD);
    k_fill_f32<<<nb(CC, T), T, 0, stream>>>(pcnt, 0.f, CC);
    k_pool<<<NN, DD, 0, stream>>>(assign, x2, psum, pcnt);
    k_poolout<<<nb(CC * DD, T), T, 0, stream>>>(psum, pcnt, out);
    k_edgeout<<<nb(2 * EE, T), T, 0, stream>>>(src, dst, assign, out);
    k_batchout<<<nb(CC, T), T, 0, stream>>>(out);
}
